// Fp8Linear_8177617732173
// MI455X (gfx1250) — 8-bit matrix operand order repaired and checked by reading, compiled, not run on hardware
//
#include <hip/hip_runtime.h>

#define M_ROWS 4096
#define K_DIM  4096
#define N_OUT  14336
#define NKB    (K_DIM / 128)
#define WQ_BYTES ((size_t)N_OUT * K_DIM)
#define XQ_BYTES ((size_t)M_ROWS * K_DIM)

typedef __attribute__((ext_vector_type(16))) int   v16i;
typedef __attribute__((ext_vector_type(8)))  float v8f;

__device__ inline unsigned int f32_to_e4m3(float x) {
    unsigned int u = __float_as_uint(x);
    unsigned int s = (u >> 24) & 0x80u;
    float a = fabsf(x);
    if (!(a < 448.0f)) return s | 0x7Eu;
    if (a < 0.015625f) {
        unsigned int m = (unsigned int)(a * 512.0f + 0.5f);
        return s | (m > 7u ? 0x08u : m);
    }
    unsigned int ua = u & 0x7FFFFFFFu;
    unsigned int lsb = (ua >> 20) & 1u;
    ua += 0x0007FFFFu + lsb;
    int e8 = (int)(ua >> 23) - 127 + 7;
    unsigned int man = (ua >> 20) & 7u;
    if (e8 > 15 || (e8 == 15 && man > 6u)) return s | 0x7Eu;
    return s | ((unsigned int)e8 << 3) | man;
}

__device__ inline unsigned int pack4_e4m3(float4 f, float sc) {
    unsigned int b0 = f32_to_e4m3(f.x * sc);
    unsigned int b1 = f32_to_e4m3(f.y * sc);
    unsigned int b2 = f32_to_e4m3(f.z * sc);
    unsigned int b3 = f32_to_e4m3(f.w * sc);
    return b0 | (b1 << 8) | (b2 << 16) | (b3 << 24);
}

__device__ inline void async_copy_b128(const void* lds_dst, const void* global_src) {
    unsigned lds_off = (unsigned)(unsigned long long)lds_dst;
    unsigned long long ga = (unsigned long long)global_src;
    asm volatile("global_load_async_to_lds_b128 %0, %1, off"
                 :: "v"(lds_off), "v"(ga)
                 : "memory");
}

__global__ __launch_bounds__(256) void quant_weight(const float* __restrict__ w,
                                                    unsigned char* __restrict__ wq) {
    size_t i = (size_t)blockIdx.x * 256 + threadIdx.x;
    if (i >= WQ_BYTES / 16) return;
    const float4* w4 = (const float4*)w;
    uint4 o;
    o.x = pack4_e4m3(w4[i * 4 + 0], 1.0f);
    o.y = pack4_e4m3(w4[i * 4 + 1], 1.0f);
    o.z = pack4_e4m3(w4[i * 4 + 2], 1.0f);
    o.w = pack4_e4m3(w4[i * 4 + 3], 1.0f);
    ((uint4*)wq)[i] = o;
}

__global__ __launch_bounds__(256) void quant_act(const float* __restrict__ x,
                                                 unsigned char* __restrict__ xq,
                                                 float* __restrict__ xs) {
    int row = blockIdx.x;
    int t = threadIdx.x;
    const float4* xr4 = (const float4*)(x + (size_t)row * K_DIM);
    float4 v[4];
    float amax = 0.0f;
#pragma unroll
    for (int i = 0; i < 4; ++i) {
        v[i] = xr4[t * 4 + i];
        amax = fmaxf(amax, fmaxf(fmaxf(fabsf(v[i].x), fabsf(v[i].y)),
                                 fmaxf(fabsf(v[i].z), fabsf(v[i].w))));
    }
#pragma unroll
    for (int off = 16; off > 0; off >>= 1)
        amax = fmaxf(amax, __shfl_xor(amax, off, 32));
    __shared__ float red[8];
    if ((t & 31) == 0) red[t >> 5] = amax;
    __syncthreads();
    float m = red[0];
#pragma unroll
    for (int i = 1; i < 8; ++i) m = fmaxf(m, red[i]);
    float scale = (m + 1e-8f) * (1.0f / 448.0f);
    float inv   = 448.0f / (m + 1e-8f);
    if (t == 0) xs[row] = scale;
    uint4 o;
    o.x = pack4_e4m3(v[0], inv);
    o.y = pack4_e4m3(v[1], inv);
    o.z = pack4_e4m3(v[2], inv);
    o.w = pack4_e4m3(v[3], inv);
    ((uint4*)(xq + (size_t)row * K_DIM))[t] = o;
}

__global__ __launch_bounds__(256) void fp8_gemm(const unsigned char* __restrict__ xq,
                                                const unsigned char* __restrict__ wq,
                                                const float* __restrict__ xs,
                                                const float* __restrict__ ws,
                                                const float* __restrict__ bias,
                                                float* __restrict__ out) {
    __shared__ uint4 Ald[2][1024];
    __shared__ uint4 Bld[2][1024];

    const int tid  = threadIdx.x;
    const int lane = tid & 31;
    const int wave = tid >> 5;
    const int bm = blockIdx.y * 128;
    const int bn = blockIdx.x * 128;

    const int K16 = K_DIM / 16;
    const uint4* Ag = (const uint4*)(xq + (size_t)bm * K_DIM);
    const uint4* Bg = (const uint4*)(wq + (size_t)bn * K_DIM);

    const int halfA = (lane & 16) ? 8 : 0;
    const int halfB = (lane & 16) ? 16 : 0;
    const int mrow  = (wave << 4) + (lane & 15);

    v8f zero = {};
    v8f acc[8];
#pragma unroll
    for (int t = 0; t < 8; ++t) acc[t] = zero;

    auto issue_tile = [&](int buf, int kb) {
#pragma unroll
        for (int i = 0; i < 4; ++i) {
            int idx = tid + i * 256;
            int r = idx >> 3, c = idx & 7;
            async_copy_b128(&Ald[buf][idx], &Ag[(size_t)r * K16 + kb * 8 + c]);
            async_copy_b128(&Bld[buf][idx], &Bg[(size_t)r * K16 + kb * 8 + c]);
        }
    };

    issue_tile(0, 0);

    for (int kb = 0; kb < NKB; ++kb) {
        const int cur = kb & 1;
        if (kb + 1 < NKB) {
            issue_tile(cur ^ 1, kb + 1);
            asm volatile("s_wait_asynccnt 0x8" ::: "memory");
        } else {
            asm volatile("s_wait_asynccnt 0x0" ::: "memory");
        }
        __syncthreads();

        v16i a;
        const int* Arow = (const int*)&Ald[cur][0] + mrow * 32;
#pragma unroll
        for (int v = 0; v < 16; ++v) {
            int v7 = v & 7;
            int ks = ((v >> 3) << 6) + ((v7 >> 2) << 5) + (((v7 >> 1) & 1) << 4)
                   + ((v7 & 1) << 2) + halfA;
            a[v] = Arow[ks >> 2];
        }

#pragma unroll
        for (int t = 0; t < 8; ++t) {
            int n = (t << 4) + (lane & 15);
            const int* Brow = (const int*)&Bld[cur][0] + n * 32;
            v16i b;
#pragma unroll
            for (int v = 0; v < 16; ++v) {
                int ks = ((v >> 3) << 6) + (((v & 7) >> 2) << 5) + (((v >> 1) & 1) << 4) + ((v & 1) << 2) + halfA;
                b[v] = Brow[ks >> 2];
            }
            acc[t] = __builtin_amdgcn_wmma_f32_16x16x128_fp8_fp8(
                a, b, (short)0, acc[t], false, false);
        }
        __syncthreads();
    }

    const int col0 = bn + (lane & 15);
#pragma unroll
    for (int t = 0; t < 8; ++t) {
        int col = col0 + (t << 4);
        float wsc = ws[col];
        float bi  = bias[col];
#pragma unroll
        for (int r = 0; r < 8; ++r) {
            int row = bm + (wave << 4) + r + halfA;
            out[(size_t)row * N_OUT + col] = acc[t][r] * xs[row] * wsc + bi;
        }
    }
}

extern "C" void kernel_launch(void* const* d_in, const int* in_sizes, int n_in,
                              void* d_out, int out_size, void* d_ws, size_t ws_size,
                              hipStream_t stream) {
    const float* x      = (const float*)d_in[0];
    const float* qw     = (const float*)d_in[1];
    const float* wscale = (const float*)d_in[2];
    const float* bias   = (const float*)d_in[3];
    float* out = (float*)d_out;

    unsigned char* wq = (unsigned char*)d_ws;
    unsigned char* xq = wq + WQ_BYTES;
    float*         xs = (float*)(xq + XQ_BYTES);

    quant_weight<<<(int)(WQ_BYTES / 16 / 256), 256, 0, stream>>>(qw, wq);
    quant_act<<<M_ROWS, 256, 0, stream>>>(x, xq, xs);

    dim3 grid(N_OUT / 128, M_ROWS / 128);
    fp8_gemm<<<grid, 256, 0, stream>>>(xq, wq, xs, wscale, bias, out);
}
